// VectorQuantizer_43894565765540
// MI455X (gfx1250) — matrix operand order repaired and checked by reading, not compiled, not run on hardware
//
#include <hip/hip_runtime.h>

typedef __attribute__((ext_vector_type(16))) __bf16 v16bf;
typedef __attribute__((ext_vector_type(8)))  float  v8f;

#define NUM_EMB 1024
#define EDIM 64
#define HW 4096            // 64*64 spatial
#define CHW 262144         // 64 * 4096 (per-batch stride in elements)
#define NPIX 131072        // 32 * 4096 pixels
#define WAVES_PER_WG 8
#define PIX_PER_WG (WAVES_PER_WG * 16)

// LDS layout (bytes)
#define S_EHI   0            // 1024*64 bf16 = 131072 B
#define S_ELO   131072       // 1024*64 bf16 = 131072 B
#define S_NRM   262144       // 1024 f32     =   4096 B
#define S_IDX   266240       // 8*16 int     =    512 B
#define S_WSUM  266752       // 8 f32        =     32 B
#define SMEM_BYTES 266784

__global__ __launch_bounds__(256) void vq_kernel(const float* __restrict__ z,
                                                 const float* __restrict__ emb,
                                                 float* __restrict__ out,
                                                 float* __restrict__ wg_loss) {
    extern __shared__ unsigned char smem[];
    __bf16* s_ehi = (__bf16*)(smem + S_EHI);
    __bf16* s_elo = (__bf16*)(smem + S_ELO);
    float*  s_nrm = (float*) (smem + S_NRM);
    int*    s_idx = (int*)   (smem + S_IDX);
    float*  s_ws  = (float*) (smem + S_WSUM);

    const int t = threadIdx.x;

    // ---- Phase 1: stage full codebook in LDS as bf16 hi/lo + half-norms ----
    for (int code = t; code < NUM_EMB; code += 256) {
        float nrm = 0.f;
        const float4* src = (const float4*)(emb + code * EDIM);
        #pragma unroll
        for (int c4 = 0; c4 < EDIM / 4; ++c4) {
            float4 v = src[c4];
            float vv[4] = {v.x, v.y, v.z, v.w};
            #pragma unroll
            for (int i = 0; i < 4; ++i) {
                float x = vv[i];
                __bf16 hi = (__bf16)x;
                __bf16 lo = (__bf16)(x - (float)hi);
                s_ehi[code * EDIM + c4 * 4 + i] = hi;
                s_elo[code * EDIM + c4 * 4 + i] = lo;
                nrm += x * x;
            }
        }
        s_nrm[code] = 0.5f * nrm;
    }
    __syncthreads();

    const int lane = t & 31;
    const int wid  = t >> 5;
    const int g    = lane >> 4;    // 16-lane group within wave
    const int col  = lane & 15;
    const int tile = (blockIdx.x * WAVES_PER_WG + wid) * 16;

    // ---- Phase 2: load A fragments (this lane's pixel, split hi/lo) ----
    // A fragment k order used here (same as B): row M = col; element e -> K = e + 16*g
    const int p     = tile + col;
    const int pb    = p >> 12;
    const int phw   = p & (HW - 1);
    const int zbase = pb * CHW + phw;

    v16bf ahi[2], alo[2];
    #pragma unroll
    for (int ch = 0; ch < 2; ++ch) {
        const int c0 = ch * 32;
        #pragma unroll
        for (int e = 0; e < 16; ++e) {
            const int k = e + 16 * g;
            float x = z[zbase + (c0 + k) * HW];
            __bf16 hi = (__bf16)x;
            ahi[ch][e] = hi;
            alo[ch][e] = (__bf16)(x - (float)hi);
        }
    }

    // ---- Phase 3: sweep all 1024 codes, 16/tile, double-buffered B + WMMA ----
    float best[8];
    int   bidx[8];
    #pragma unroll
    for (int j = 0; j < 8; ++j) { best[j] = -3.0e38f; bidx[j] = 0; }

    const int bcol = col * EDIM + 16 * g;   // this lane's column base in codebook rows

    // B fragment buffers (double-buffered across tile iterations)
    v16bf bhA0, bhA1, blA0, blA1; float hnA;
    v16bf bhB0, bhB1, blB0, blB1; float hnB;

#define LOAD_B(KT, BH0, BH1, BL0, BL1, HN) do {                              \
        const int code_ = (KT) * 16 + col;                                   \
        const __bf16* hp_ = s_ehi + (KT) * (16 * EDIM) + bcol;               \
        const __bf16* lp_ = s_elo + (KT) * (16 * EDIM) + bcol;               \
        BH0 = *(const v16bf*)(hp_);                                          \
        BH1 = *(const v16bf*)(hp_ + 32);                                     \
        BL0 = *(const v16bf*)(lp_);                                          \
        BL1 = *(const v16bf*)(lp_ + 32);                                     \
        HN  = s_nrm[code_];                                                  \
    } while (0)

#define PROC_B(KT, BH0, BH1, BL0, BL1, HN) do {                              \
        v8f acc = {};                                                        \
        acc = __builtin_amdgcn_wmma_f32_16x16x32_bf16(false, ahi[0], false,  \
                    BH0, (short)0, acc, false, false);                       \
        acc = __builtin_amdgcn_wmma_f32_16x16x32_bf16(false, ahi[0], false,  \
                    BL0, (short)0, acc, false, false);                       \
        acc = __builtin_amdgcn_wmma_f32_16x16x32_bf16(false, alo[0], false,  \
                    BH0, (short)0, acc, false, false);                       \
        acc = __builtin_amdgcn_wmma_f32_16x16x32_bf16(false, ahi[1], false,  \
                    BH1, (short)0, acc, false, false);                       \
        acc = __builtin_amdgcn_wmma_f32_16x16x32_bf16(false, ahi[1], false,  \
                    BL1, (short)0, acc, false, false);                       \
        acc = __builtin_amdgcn_wmma_f32_16x16x32_bf16(false, alo[1], false,  \
                    BH1, (short)0, acc, false, false);                       \
        const int code_ = (KT) * 16 + col;                                   \
        _Pragma("unroll")                                                    \
        for (int j = 0; j < 8; ++j) {                                        \
            float s_ = acc[j] - HN;            /* argmax(z.e - 0.5|e|^2) */  \
            if (s_ > best[j]) { best[j] = s_; bidx[j] = code_; }             \
        }                                                                    \
    } while (0)

    LOAD_B(0, bhA0, bhA1, blA0, blA1, hnA);
    for (int kt = 0; kt < NUM_EMB / 16; kt += 2) {
        // prefetch tile kt+1 while computing tile kt
        LOAD_B(kt + 1, bhB0, bhB1, blB0, blB1, hnB);
        PROC_B(kt, bhA0, bhA1, blA0, blA1, hnA);
        // prefetch tile kt+2 (clamped; final redundant load is harmless)
        const int kn = (kt + 2 < NUM_EMB / 16) ? kt + 2 : (NUM_EMB / 16 - 1);
        LOAD_B(kn, bhA0, bhA1, blA0, blA1, hnA);
        PROC_B(kt + 1, bhB0, bhB1, blB0, blB1, hnB);
    }
#undef LOAD_B
#undef PROC_B

    // ---- Phase 4: cross-lane argmax within each 16-lane column group ----
    #pragma unroll
    for (int j = 0; j < 8; ++j) {
        float s = best[j];
        int   bi = bidx[j];
        #pragma unroll
        for (int m = 1; m <= 8; m <<= 1) {
            float os = __shfl_xor(s, m, 32);
            int   oi = __shfl_xor(bi, m, 32);
            if (os > s || (os == s && oi < bi)) { s = os; bi = oi; }
        }
        // slot j of group g is pixel row j + 8*g
        if (col == j) s_idx[wid * 16 + 8 * g + j] = bi;
    }
    // Per-wave LDS ops are in-order (DScnt); safe to read back within the wave.

    // ---- Phase 5: gather codeword, write out in [B,C,H,W], accumulate loss ----
    float lsum = 0.f;
    {
        const int r     = col;
        const int pr    = tile + r;
        const int rb    = pr >> 12;
        const int rhw   = pr & (HW - 1);
        const int obase = rb * CHW + rhw;
        const int myidx = s_idx[wid * 16 + r];
        const float* erow = emb + myidx * EDIM;
        #pragma unroll
        for (int cc = 0; cc < 32; ++cc) {
            const int c = 2 * cc + g;                 // lanes r and r+16 split even/odd c
            float e  = erow[c];
            float zv = z[obase + c * HW];
            out[obase + c * HW] = e;
            float d = e - zv;
            lsum += d * d;
        }
    }

    // ---- Phase 6: deterministic loss partials per workgroup ----
    #pragma unroll
    for (int m = 1; m < 32; m <<= 1) lsum += __shfl_xor(lsum, m, 32);
    if (lane == 0) s_ws[wid] = lsum;
    __syncthreads();
    if (t == 0) {
        float tot = 0.f;
        #pragma unroll
        for (int w = 0; w < WAVES_PER_WG; ++w) tot += s_ws[w];
        wg_loss[blockIdx.x] = tot;
    }
}

__global__ __launch_bounds__(256) void vq_loss_reduce(const float* __restrict__ ws,
                                                      float* __restrict__ loss_out) {
    __shared__ float sm[256];
    const int t = threadIdx.x;
    float s = 0.f;
    #pragma unroll
    for (int i = 0; i < 4; ++i) s += ws[t + 256 * i];
    sm[t] = s;
    __syncthreads();
    for (int off = 128; off > 0; off >>= 1) {
        if (t < off) sm[t] += sm[t + off];
        __syncthreads();
    }
    // loss = mse + 0.5*mse = 1.5 * sum / (N*C)
    if (t == 0) loss_out[0] = sm[0] * (1.5f / 8388608.0f);
}

extern "C" void kernel_launch(void* const* d_in, const int* in_sizes, int n_in,
                              void* d_out, int out_size, void* d_ws, size_t ws_size,
                              hipStream_t stream) {
    const float* z   = (const float*)d_in[0];
    const float* emb = (const float*)d_in[1];
    float* out = (float*)d_out;
    float* ws  = (float*)d_ws;   // 1024 partial-loss floats

    const int num_wg = NPIX / PIX_PER_WG;   // 1024
    vq_kernel<<<dim3(num_wg), dim3(256), SMEM_BYTES, stream>>>(z, emb, out, ws);
    vq_loss_reduce<<<dim3(1), dim3(256), 0, stream>>>(ws, out + (out_size - 1));
}
